// ReTransformerEncoderLayer_7421703487594
// MI455X (gfx1250) — compile-verified, matrix operand order checked by reading, not run on hardware
//
#include <hip/hip_runtime.h>
#include <hip/hip_bf16.h>

// ---------------------------------------------------------------------------
// Problem constants (from reference)
// ---------------------------------------------------------------------------
#define DMODEL 1024
#define NHEADS 16
#define HDIM   64
#define SEQ    1024
#define BATCH  4
#define MROWS  (BATCH * SEQ)   // 4096 token rows
#define DFF    (4 * DMODEL)    // 4096

// ---------------------------------------------------------------------------
// WMMA types (CDNA5 wave32: 16x16x32 bf16 -> f32)
// ---------------------------------------------------------------------------
typedef __attribute__((ext_vector_type(16))) __bf16 v16bf;
typedef __attribute__((ext_vector_type(8)))  float  v8f;

union Frag16 {
    v16bf v;
    uint4 q[2];   // two 16B chunks = 16 bf16 per lane
};

static __device__ __forceinline__ unsigned short f2bf(float f) {
    unsigned int u = __float_as_uint(f);
    u += 0x7FFFu + ((u >> 16) & 1u);      // round-to-nearest-even
    return (unsigned short)(u >> 16);
}

// ---------------------------------------------------------------------------
// f32 -> bf16 cast
// ---------------------------------------------------------------------------
__global__ __launch_bounds__(256) void cvt_f32_bf16(const float* __restrict__ in,
                                                    unsigned short* __restrict__ out,
                                                    int n) {
    int i = blockIdx.x * 256 + threadIdx.x;
    if (i < n) out[i] = f2bf(in[i]);
}

// ---------------------------------------------------------------------------
// Per-head transpose: Xt[n][h][d][s] = Ab[n*SEQ+s][h*HDIM+d]
// gives contiguous K-rows for the P@V GEMM's B operand (V^T)
// ---------------------------------------------------------------------------
__global__ __launch_bounds__(256) void head_transpose(const unsigned short* __restrict__ Ab,
                                                      unsigned short* __restrict__ Xt) {
    int idx = blockIdx.x * 256 + threadIdx.x;         // over 4M elements
    int s = idx & (SEQ - 1);
    int t = idx >> 10;
    int d = t & (HDIM - 1); t >>= 6;
    int h = t & (NHEADS - 1);
    int n = t >> 4;
    Xt[idx] = Ab[((size_t)(n * SEQ + s)) * DMODEL + h * HDIM + d];
}

// ---------------------------------------------------------------------------
// Attention (per reference: no qkv projections, no 1/sqrt(d) scaling).
// One block = (batch*head, 32 query rows). Scores kept f32 in LDS (128KB),
// probabilities bf16 in LDS (64KB) -> 192KB of the WGP's 320KB LDS.
// ---------------------------------------------------------------------------
__global__ __launch_bounds__(256) void attn_kernel(const unsigned short* __restrict__ Ab,
                                                   const unsigned short* __restrict__ Xt,
                                                   unsigned short* __restrict__ Cb) {
    __shared__ __align__(16) float          sS[32 * SEQ];
    __shared__ __align__(16) unsigned short sP[32 * SEQ];

    const int wave = threadIdx.x >> 5;
    const int lane = threadIdx.x & 31;
    const int half = lane >> 4;
    const int l16  = lane & 15;
    const int qb = blockIdx.x * 32;
    const int nh = blockIdx.y;              // n*NHEADS + h
    const int nb = nh >> 4;
    const int h  = nh & (NHEADS - 1);

    const unsigned short* base = Ab + (size_t)nb * SEQ * DMODEL + h * HDIM;

    // ---- preload Q fragments (2 M-tiles x 2 K-steps over HDIM=64) ----
    Frag16 qf[2][2];
#pragma unroll
    for (int mt = 0; mt < 2; ++mt)
#pragma unroll
        for (int ks = 0; ks < 2; ++ks) {
            const unsigned short* qp =
                base + (size_t)(qb + mt * 16 + l16) * DMODEL + ks * 32 + half * 8;
            qf[mt][ks].q[0] = *(const uint4*)qp;
            qf[mt][ks].q[1] = *(const uint4*)(qp + 16);
        }

    // ---- scores = Q @ K^T : each wave owns 8 key tiles of width 16 ----
    for (int nt = wave * 8; nt < wave * 8 + 8; ++nt) {
        v8f acc0 = {}; v8f acc1 = {};
#pragma unroll
        for (int ks = 0; ks < 2; ++ks) {
            Frag16 kf;
            const unsigned short* kp =
                base + (size_t)(nt * 16 + l16) * DMODEL + ks * 32 + half * 8;
            kf.q[0] = *(const uint4*)kp;
            kf.q[1] = *(const uint4*)(kp + 16);
            acc0 = __builtin_amdgcn_wmma_f32_16x16x32_bf16(
                false, qf[0][ks].v, false, kf.v, (short)0, acc0, false, false);
            acc1 = __builtin_amdgcn_wmma_f32_16x16x32_bf16(
                false, qf[1][ks].v, false, kf.v, (short)0, acc1, false, false);
        }
        const int col = nt * 16 + l16;
#pragma unroll
        for (int e = 0; e < 8; ++e) {
            sS[(e + 8 * half) * SEQ + col]      = acc0[e];
            sS[(16 + e + 8 * half) * SEQ + col] = acc1[e];
        }
    }
    __syncthreads();

    // ---- softmax over keys: 8 consecutive lanes per query row ----
    {
        const int row = threadIdx.x >> 3;   // 0..31
        const int sub = threadIdx.x & 7;
        float* srow = &sS[row * SEQ];
        float mx = -3.0e38f;
        for (int c = sub; c < SEQ; c += 8) mx = fmaxf(mx, srow[c]);
#pragma unroll
        for (int o = 1; o < 8; o <<= 1) mx = fmaxf(mx, __shfl_xor(mx, o));
        float sum = 0.f;
        for (int c = sub; c < SEQ; c += 8) sum += __expf(srow[c] - mx);
#pragma unroll
        for (int o = 1; o < 8; o <<= 1) sum += __shfl_xor(sum, o);
        const float inv = 1.0f / sum;
        unsigned short* prow = &sP[row * SEQ];
        for (int c = sub; c < SEQ; c += 8) prow[c] = f2bf(__expf(srow[c] - mx) * inv);
    }
    __syncthreads();

    // ---- out = P @ V : 2 M-tiles x 4 N-tiles, one 16x16 tile per wave ----
    {
        const int mt = wave >> 2;
        const int nt = wave & 3;
        const unsigned short* vt = Xt + ((size_t)nh * HDIM + nt * 16 + l16) * SEQ;
        v8f acc = {};
        for (int kk = 0; kk < SEQ; kk += 32) {
            Frag16 pf, vf;
            const unsigned short* pp = &sP[(mt * 16 + l16) * SEQ + kk + half * 8];
            pf.q[0] = *(const uint4*)pp;             // ds_load_b128
            pf.q[1] = *(const uint4*)(pp + 16);
            const unsigned short* vp = vt + kk + half * 8;
            vf.q[0] = *(const uint4*)vp;
            vf.q[1] = *(const uint4*)(vp + 16);
            acc = __builtin_amdgcn_wmma_f32_16x16x32_bf16(
                false, pf.v, false, vf.v, (short)0, acc, false, false);
        }
        const int d = nt * 16 + l16;
#pragma unroll
        for (int e = 0; e < 8; ++e) {
            const int q = qb + mt * 16 + e + 8 * half;
            Cb[((size_t)(nb * SEQ + q)) * DMODEL + h * HDIM + d] = f2bf(acc[e]);
        }
    }
}

// ---------------------------------------------------------------------------
// Generic bf16 GEMM: C[M,N] = A[M,K] @ W[N,K]^T + bias, optional ReLU,
// output f32 (Cf) or bf16 (Cbo). Block tile 128x64, 8 waves, 32x32/wave.
// ---------------------------------------------------------------------------
__global__ __launch_bounds__(256) void gemm_bf16(const unsigned short* __restrict__ A,
                                                 const unsigned short* __restrict__ W,
                                                 const float* __restrict__ bias,
                                                 float* __restrict__ Cf,
                                                 unsigned short* __restrict__ Cbo,
                                                 int M, int N, int K, int relu) {
    const int wave = threadIdx.x >> 5;
    const int lane = threadIdx.x & 31;
    const int half = lane >> 4;
    const int l16  = lane & 15;

    const int wm = blockIdx.x * 128 + (wave >> 1) * 32;
    const int wn = blockIdx.y * 64  + (wave & 1) * 32;

    v8f acc[2][2] = {{{}, {}}, {{}, {}}};

    for (int k0 = 0; k0 < K; k0 += 32) {
        Frag16 a[2], b[2];
#pragma unroll
        for (int i = 0; i < 2; ++i) {
            const unsigned short* ap = A + (size_t)(wm + i * 16 + l16) * K + k0 + half * 8;
            a[i].q[0] = *(const uint4*)ap;
            a[i].q[1] = *(const uint4*)(ap + 16);
        }
#pragma unroll
        for (int j = 0; j < 2; ++j) {
            const unsigned short* bp = W + (size_t)(wn + j * 16 + l16) * K + k0 + half * 8;
            b[j].q[0] = *(const uint4*)bp;
            b[j].q[1] = *(const uint4*)(bp + 16);
        }
#pragma unroll
        for (int i = 0; i < 2; ++i)
#pragma unroll
            for (int j = 0; j < 2; ++j)
                acc[i][j] = __builtin_amdgcn_wmma_f32_16x16x32_bf16(
                    false, a[i].v, false, b[j].v, (short)0, acc[i][j], false, false);
    }

#pragma unroll
    for (int j = 0; j < 2; ++j) {
        const int col = wn + j * 16 + l16;
        const float bv = bias[col];
#pragma unroll
        for (int i = 0; i < 2; ++i) {
#pragma unroll
            for (int e = 0; e < 8; ++e) {
                const int row = wm + i * 16 + e + 8 * half;
                float v = acc[i][j][e] + bv;
                if (relu) v = fmaxf(v, 0.0f);
                if (Cbo) Cbo[(size_t)row * N + col] = f2bf(v);
                else     Cf[(size_t)row * N + col]  = v;
            }
        }
    }
}

// ---------------------------------------------------------------------------
// Fused residual add + LayerNorm. One block per token row. Optionally also
// emits the bf16 copy consumed by the next GEMM.
// ---------------------------------------------------------------------------
__global__ __launch_bounds__(256) void add_ln(const float* __restrict__ X,
                                              const float* __restrict__ R,
                                              const float* __restrict__ g,
                                              const float* __restrict__ b,
                                              float* __restrict__ Yf,
                                              unsigned short* __restrict__ Ybf) {
    __shared__ float rs[8], rs2[8];
    const int row = blockIdx.x;
    const float* xr = X + (size_t)row * DMODEL;
    const float* rr = R + (size_t)row * DMODEL;

    float s = 0.f, s2 = 0.f;
    for (int c = threadIdx.x; c < DMODEL; c += 256) {
        const float v = xr[c] + rr[c];
        s += v; s2 += v * v;
    }
#pragma unroll
    for (int o = 16; o > 0; o >>= 1) {
        s  += __shfl_down(s, o);
        s2 += __shfl_down(s2, o);
    }
    if ((threadIdx.x & 31) == 0) { rs[threadIdx.x >> 5] = s; rs2[threadIdx.x >> 5] = s2; }
    __syncthreads();
    float ts = 0.f, ts2 = 0.f;
#pragma unroll
    for (int i = 0; i < 8; ++i) { ts += rs[i]; ts2 += rs2[i]; }
    const float mu   = ts * (1.0f / DMODEL);
    const float var  = ts2 * (1.0f / DMODEL) - mu * mu;
    const float rstd = rsqrtf(var + 1e-5f);

    for (int c = threadIdx.x; c < DMODEL; c += 256) {
        const float v = (xr[c] + rr[c] - mu) * rstd * g[c] + b[c];
        Yf[(size_t)row * DMODEL + c] = v;
        if (Ybf) Ybf[(size_t)row * DMODEL + c] = f2bf(v);
    }
}

// ---------------------------------------------------------------------------
// Launcher: layer_num is the static python int 2 in the reference setup, so
// the executed path is LN(x+attn1) -> LN(+attn2) -> FFN -> LN(+ff).
// ---------------------------------------------------------------------------
extern "C" void kernel_launch(void* const* d_in, const int* in_sizes, int n_in,
                              void* d_out, int out_size, void* d_ws, size_t ws_size,
                              hipStream_t stream) {
    const float* x   = (const float*)d_in[0];
    const float* Wo1 = (const float*)d_in[1];
    const float* bo1 = (const float*)d_in[2];
    const float* Wo2 = (const float*)d_in[3];
    const float* bo2 = (const float*)d_in[4];
    const float* W1  = (const float*)d_in[5];
    const float* b1  = (const float*)d_in[6];
    const float* W2  = (const float*)d_in[7];
    const float* b2  = (const float*)d_in[8];
    const float* g1  = (const float*)d_in[9];
    const float* be1 = (const float*)d_in[10];
    const float* g2  = (const float*)d_in[11];
    const float* be2 = (const float*)d_in[12];
    const float* g3  = (const float*)d_in[13];
    const float* be3 = (const float*)d_in[14];
    float* out = (float*)d_out;

    char* ws = (char*)d_ws;
    unsigned short* A_bf = (unsigned short*)(ws);                          //  8 MB activation bf16
    unsigned short* Wt   = (unsigned short*)(ws + ((size_t)8  << 20));     //  8 MB weight staging bf16
    unsigned short* Xt   = (unsigned short*)(ws + ((size_t)16 << 20));     //  8 MB V^T bf16
    unsigned short* Cb   = (unsigned short*)(ws + ((size_t)24 << 20));     //  8 MB attn context bf16
    float*          F0   = (float*)(ws + ((size_t)32 << 20));              // 16 MB
    float*          F1   = (float*)(ws + ((size_t)48 << 20));              // 16 MB
    float*          F2   = (float*)(ws + ((size_t)64 << 20));              // 16 MB
    unsigned short* Hb   = (unsigned short*)(ws + ((size_t)80 << 20));     // 32 MB FFN hidden bf16

    const dim3 blk(256);
    const int nAct = MROWS * DMODEL;          // 4M
    const int nWo  = DMODEL * DMODEL;         // 1M
    const int nW1  = DFF * DMODEL;            // 4M

    // ---- attn1 = MHA(x) @ Wo1^T + bo1 -> F1 ----
    cvt_f32_bf16<<<nAct / 256, blk, 0, stream>>>(x, A_bf, nAct);
    head_transpose<<<nAct / 256, blk, 0, stream>>>(A_bf, Xt);
    attn_kernel<<<dim3(SEQ / 32, BATCH * NHEADS), blk, 0, stream>>>(A_bf, Xt, Cb);
    cvt_f32_bf16<<<nWo / 256, blk, 0, stream>>>(Wo1, Wt, nWo);
    gemm_bf16<<<dim3(MROWS / 128, DMODEL / 64), blk, 0, stream>>>(
        Cb, Wt, bo1, F1, nullptr, MROWS, DMODEL, DMODEL, 0);

    // ---- attn2 = MHA(attn1) @ Wo2^T + bo2 -> F2 ----
    cvt_f32_bf16<<<nAct / 256, blk, 0, stream>>>(F1, A_bf, nAct);
    head_transpose<<<nAct / 256, blk, 0, stream>>>(A_bf, Xt);
    attn_kernel<<<dim3(SEQ / 32, BATCH * NHEADS), blk, 0, stream>>>(A_bf, Xt, Cb);
    cvt_f32_bf16<<<nWo / 256, blk, 0, stream>>>(Wo2, Wt, nWo);
    gemm_bf16<<<dim3(MROWS / 128, DMODEL / 64), blk, 0, stream>>>(
        Cb, Wt, bo2, F2, nullptr, MROWS, DMODEL, DMODEL, 0);

    // ---- x = LN(x + attn1) -> F0 ; x = LN(F0 + attn2) -> F1 (+bf16) ----
    add_ln<<<MROWS, blk, 0, stream>>>(x,  F1, g1, be1, F0, nullptr);
    add_ln<<<MROWS, blk, 0, stream>>>(F0, F2, g2, be2, F1, A_bf);

    // ---- ff = relu(x @ W1^T + b1) @ W2^T + b2 -> F2 ----
    cvt_f32_bf16<<<nW1 / 256, blk, 0, stream>>>(W1, Wt, nW1);
    gemm_bf16<<<dim3(MROWS / 128, DFF / 64), blk, 0, stream>>>(
        A_bf, Wt, b1, nullptr, Hb, MROWS, DFF, DMODEL, 1);
    cvt_f32_bf16<<<nW1 / 256, blk, 0, stream>>>(W2, Wt, nW1);
    gemm_bf16<<<dim3(MROWS / 128, DMODEL / 64), blk, 0, stream>>>(
        Hb, Wt, b2, F2, nullptr, MROWS, DMODEL, DFF, 0);

    // ---- out = LN(x + ff) ----
    add_ln<<<MROWS, blk, 0, stream>>>(F1, F2, g3, be3, out, nullptr);
}
